// Head_16509854286018
// MI455X (gfx1250) — matrix operand order repaired and checked by reading, not compiled, not run on hardware
//
#include <hip/hip_runtime.h>
#include <hip/hip_bf16.h>

#define EMBED  768
#define HEADSZ 64
#define SEQ    2048
#define BATCH  4

typedef _Float16 h16;
typedef __attribute__((ext_vector_type(16))) _Float16 v16h;
typedef __attribute__((ext_vector_type(8)))  _Float16 v8h;
typedef __attribute__((ext_vector_type(8)))  float    v8f;
typedef __attribute__((ext_vector_type(4)))  float    v4f;

union U16h { v16h v; v8h h[2]; };

__device__ __forceinline__ v8f wmma_f16(v16h a, v16h b, v8f c) {
  // D = A(16x32 f16) x B(32x16 f16) + C(16x16 f32)
  return __builtin_amdgcn_wmma_f32_16x16x32_f16(false, a, false, b, (short)0, c, false, false);
}

__device__ __forceinline__ v8f vzero8() {
  v8f z;
#pragma unroll
  for (int i = 0; i < 8; ++i) z[i] = 0.0f;
  return z;
}

// ---------------------------------------------------------------------------
// Kernel 0: W [E][H] fp32  ->  WT [H][E] f16   (x3 matrices)
// ---------------------------------------------------------------------------
__global__ void wt_convert_kernel(const float* __restrict__ Wk,
                                  const float* __restrict__ Wq,
                                  const float* __restrict__ Wv,
                                  h16* __restrict__ WTk,
                                  h16* __restrict__ WTq,
                                  h16* __restrict__ WTv) {
  int idx = blockIdx.x * blockDim.x + threadIdx.x;
  const int per = EMBED * HEADSZ;
  if (idx >= 3 * per) return;
  int m   = idx / per;
  int rem = idx - m * per;
  int h   = rem / EMBED;
  int e   = rem - h * EMBED;
  const float* W = (m == 0) ? Wk : (m == 1) ? Wq : Wv;
  h16*        WT = (m == 0) ? WTk : (m == 1) ? WTq : WTv;
  WT[(size_t)h * EMBED + e] = (h16)W[(size_t)e * HEADSZ + h];
}

// ---------------------------------------------------------------------------
// Kernel 1: projections. One wave = 16 rows of x -> 16x64 K,Q,V tiles.
//   B-tile loads are double-buffered across the 3 weight matrices so the
//   WMMAs of matrix m overlap the loads of matrix m+1.
// ---------------------------------------------------------------------------
__global__ __launch_bounds__(128) void proj_kernel(const float* __restrict__ x,
                                                   const h16* __restrict__ WTk,
                                                   const h16* __restrict__ WTq,
                                                   const h16* __restrict__ WTv,
                                                   h16* __restrict__ Kh,
                                                   h16* __restrict__ Qh,
                                                   h16* __restrict__ Vt) {
  const int lane = threadIdx.x & 31;
  const int wave = threadIdx.x >> 5;
  const int n    = lane & 15;     // col within 16-wide tile
  const int g    = lane >> 4;     // lane group
  const int rowbase = (blockIdx.x * 4 + wave) * 16;   // 16 rows per wave

  v8f acc[3][4];
#pragma unroll
  for (int m = 0; m < 3; ++m)
#pragma unroll
    for (int nt = 0; nt < 4; ++nt) acc[m][nt] = vzero8();

  const h16* WT[3] = { WTk, WTq, WTv };
  const float* xrow = x + (size_t)(rowbase + n) * EMBED;   // A row m = lane&15

  for (int k0 = 0; k0 < EMBED; k0 += 32) {
    // ---- A-tile loads (f32) issued first ----
    const float* pa0 = xrow + k0 + 8 * g;
    const float* pa1 = xrow + k0 + 16 + 8 * g;
    v4f f0 = *(const v4f*)(pa0);
    v4f f1 = *(const v4f*)(pa0 + 4);
    v4f f2 = *(const v4f*)(pa1);
    v4f f3 = *(const v4f*)(pa1 + 4);

    // ---- prime B double-buffer with matrix 0's four tiles ----
    U16h bt[2][4];
#pragma unroll
    for (int nt = 0; nt < 4; ++nt) {
      const h16* wp = WT[0] + (size_t)(nt * 16 + n) * EMBED + k0;
      bt[0][nt].h[0] = *(const v8h*)(wp + 8 * g);
      bt[0][nt].h[1] = *(const v8h*)(wp + 16 + 8 * g);
    }

    // ---- convert A while B loads are in flight ----
    U16h a;
    {
      v8h h0, h1;
#pragma unroll
      for (int i = 0; i < 4; ++i) {
        h0[i] = (h16)f0[i]; h0[4 + i] = (h16)f1[i];
        h1[i] = (h16)f2[i]; h1[4 + i] = (h16)f3[i];
      }
      a.h[0] = h0; a.h[1] = h1;
    }

#pragma unroll
    for (int m = 0; m < 3; ++m) {
      // issue next matrix's loads before consuming the current buffer
      if (m < 2) {
#pragma unroll
        for (int nt = 0; nt < 4; ++nt) {
          const h16* wp = WT[m + 1] + (size_t)(nt * 16 + n) * EMBED + k0;
          bt[(m + 1) & 1][nt].h[0] = *(const v8h*)(wp + 8 * g);
          bt[(m + 1) & 1][nt].h[1] = *(const v8h*)(wp + 16 + 8 * g);
        }
      }
#pragma unroll
      for (int nt = 0; nt < 4; ++nt)
        acc[m][nt] = wmma_f16(a.v, bt[m & 1][nt].v, acc[m][nt]);
    }
  }

  // D layout: lane l, vgpr r -> (row = r + 8g, col = nt*16 + (l&15))
  const int b     = rowbase >> 11;       // / SEQ
  const int tbase = rowbase & (SEQ - 1);
#pragma unroll
  for (int r = 0; r < 8; ++r) {
    const int row = rowbase + r + 8 * g;
#pragma unroll
    for (int nt = 0; nt < 4; ++nt) {
      const int hc = nt * 16 + n;
      Kh[(size_t)row * HEADSZ + hc] = (h16)acc[0][nt][r];
      Qh[(size_t)row * HEADSZ + hc] = (h16)acc[1][nt][r];
      Vt[((size_t)b * HEADSZ + hc) * SEQ + (tbase + r + 8 * g)] = (h16)acc[2][nt][r];
    }
  }
}

// ---------------------------------------------------------------------------
// Kernel 2: causal flash attention. One wave = 16 query rows, loop over
// 32-key blocks. K AND V tiles are loaded up-front each block so V latency
// hides under the S WMMAs + softmax; next block is prefetched.
// ---------------------------------------------------------------------------
__global__ __launch_bounds__(128) void attn_kernel(const h16* __restrict__ Qh,
                                                   const h16* __restrict__ Kh,
                                                   const h16* __restrict__ Vt,
                                                   float* __restrict__ out) {
  __shared__ _Float16 Plds[4][16][32];   // per-wave P staging (16 rows x 32 keys)

  const int lane = threadIdx.x & 31;
  const int wave = threadIdx.x >> 5;
  const int n    = lane & 15;
  const int g    = lane >> 4;

  const int tile = blockIdx.x * 4 + wave;       // 0 .. 511
  const int b    = tile >> 7;                   // / (SEQ/16)
  const int t0   = (tile & 127) << 4;

  const h16* Qb = Qh + (size_t)b * SEQ * HEADSZ;
  const h16* Kb = Kh + (size_t)b * SEQ * HEADSZ;
  const h16* Vb = Vt + (size_t)b * HEADSZ * SEQ;

  // Q A-tiles for the two K-chunks (h = 0..31, 32..63), kept resident
  v16h qa[2];
#pragma unroll
  for (int c = 0; c < 2; ++c) {
    const h16* qp = Qb + (size_t)(t0 + n) * HEADSZ + c * 32;
    U16h u;
    u.h[0] = *(const v8h*)(qp + 8 * g);
    u.h[1] = *(const v8h*)(qp + 16 + 8 * g);
    qa[c] = u.v;
  }

  v8f oacc[4];
#pragma unroll
  for (int nt = 0; nt < 4; ++nt) oacc[nt] = vzero8();
  float mrun[8], lrun[8];
#pragma unroll
  for (int r = 0; r < 8; ++r) { mrun[r] = -__builtin_inff(); lrun[r] = 0.0f; }

  const float scale = 0.03608439182435161f;   // 1/sqrt(768)

  for (int s0 = 0; s0 <= t0 + 15; s0 += 32) {
    // ---- issue ALL loads for this block: K tiles (consumed first), V tiles ----
    U16h kt[4];   // [c*2+nt]
#pragma unroll
    for (int c = 0; c < 2; ++c) {
#pragma unroll
      for (int nt = 0; nt < 2; ++nt) {
        const h16* kp = Kb + (size_t)(s0 + nt * 16 + n) * HEADSZ + c * 32;
        kt[c * 2 + nt].h[0] = *(const v8h*)(kp + 8 * g);
        kt[c * 2 + nt].h[1] = *(const v8h*)(kp + 16 + 8 * g);
      }
    }
    U16h vt[4];
#pragma unroll
    for (int nt = 0; nt < 4; ++nt) {
      const h16* vp = Vb + (size_t)(nt * 16 + n) * SEQ + s0;
      vt[nt].h[0] = *(const v8h*)(vp + 8 * g);
      vt[nt].h[1] = *(const v8h*)(vp + 16 + 8 * g);
    }
    // prefetch next key block's K/V cachelines (speculative; guarded in-range)
    if (s0 + 32 <= t0 + 15) {
      __builtin_prefetch(Kb + (size_t)(s0 + 32 + n) * HEADSZ, 0, 1);
      __builtin_prefetch(Vb + (size_t)(g * 16 + n) * SEQ + s0 + 32, 0, 1);
    }

    // ---- S = Q K^T for keys [s0, s0+32) : two N-tiles of 16 keys ----
    v8f sacc[2] = { vzero8(), vzero8() };
#pragma unroll
    for (int c = 0; c < 2; ++c) {
#pragma unroll
      for (int nt = 0; nt < 2; ++nt)
        sacc[nt] = wmma_f16(qa[c], kt[c * 2 + nt].v, sacc[nt]);
    }

    // ---- mask -> -inf, scale, online softmax (rows live in 16-lane groups) ----
#pragma unroll
    for (int r = 0; r < 8; ++r) {
      const int qrow = t0 + r + 8 * g;
      float a0 = (s0 + n      <= qrow) ? sacc[0][r] * scale : -__builtin_inff();
      float a1 = (s0 + 16 + n <= qrow) ? sacc[1][r] * scale : -__builtin_inff();
      float mx = fmaxf(a0, a1);
      mx = fmaxf(mx, __shfl_xor(mx, 1, 32));
      mx = fmaxf(mx, __shfl_xor(mx, 2, 32));
      mx = fmaxf(mx, __shfl_xor(mx, 4, 32));
      mx = fmaxf(mx, __shfl_xor(mx, 8, 32));
      const float mnew = fmaxf(mrun[r], mx);
      const float corr = __expf(mrun[r] - mnew);
      const float p0 = __expf(a0 - mnew);
      const float p1 = __expf(a1 - mnew);
      float sum = p0 + p1;
      sum += __shfl_xor(sum, 1, 32);
      sum += __shfl_xor(sum, 2, 32);
      sum += __shfl_xor(sum, 4, 32);
      sum += __shfl_xor(sum, 8, 32);
      lrun[r] = lrun[r] * corr + sum;
      mrun[r] = mnew;
#pragma unroll
      for (int nt = 0; nt < 4; ++nt) oacc[nt][r] *= corr;
      Plds[wave][r + 8 * g][n]      = (h16)p0;
      Plds[wave][r + 8 * g][16 + n] = (h16)p1;
    }
    asm volatile("s_wait_dscnt 0x0" ::: "memory");

    // ---- reload P in A-matrix layout: row m = lane&15, two 8-half chunks ----
    U16h pu;
    pu.h[0] = *(const v8h*)&Plds[wave][n][8 * g];
    pu.h[1] = *(const v8h*)&Plds[wave][n][16 + 8 * g];

    // ---- O += P V  (V tiles already resident) ----
#pragma unroll
    for (int nt = 0; nt < 4; ++nt)
      oacc[nt] = wmma_f16(pu.v, vt[nt].v, oacc[nt]);
    asm volatile("s_wait_dscnt 0x0" ::: "memory");
  }

  // ---- normalize and store ----
#pragma unroll
  for (int r = 0; r < 8; ++r) {
    const float invl = 1.0f / lrun[r];
    float* op = out + ((size_t)b * SEQ + (t0 + r + 8 * g)) * HEADSZ;
#pragma unroll
    for (int nt = 0; nt < 4; ++nt) op[nt * 16 + n] = oacc[nt][r] * invl;
  }
}

// ---------------------------------------------------------------------------
extern "C" void kernel_launch(void* const* d_in, const int* in_sizes, int n_in,
                              void* d_out, int out_size, void* d_ws, size_t ws_size,
                              hipStream_t stream) {
  const float* x  = (const float*)d_in[0];
  const float* Wk = (const float*)d_in[1];
  const float* Wq = (const float*)d_in[2];
  const float* Wv = (const float*)d_in[3];
  float* out = (float*)d_out;

  char* ws = (char*)d_ws;
  const size_t wtBytes  = (size_t)HEADSZ * EMBED * sizeof(h16);          // 96 KB each
  const size_t qkvBytes = (size_t)BATCH * SEQ * HEADSZ * sizeof(h16);    // 1 MB each
  h16* WTk = (h16*)(ws);
  h16* WTq = (h16*)(ws + wtBytes);
  h16* WTv = (h16*)(ws + 2 * wtBytes);
  h16* Kh  = (h16*)(ws + 3 * wtBytes);
  h16* Qh  = (h16*)(ws + 3 * wtBytes + qkvBytes);
  h16* Vt  = (h16*)(ws + 3 * wtBytes + 2 * qkvBytes);

  // Kernel 0: weight transpose + f16 convert
  {
    const int total = 3 * EMBED * HEADSZ;
    wt_convert_kernel<<<(total + 255) / 256, 256, 0, stream>>>(Wk, Wq, Wv, WTk, WTq, WTv);
  }
  // Kernel 1: projections (8192 rows / 16 per wave / 4 waves per block)
  {
    const int blocks = (BATCH * SEQ) / (16 * 4);   // 128
    proj_kernel<<<blocks, 128, 0, stream>>>(x, WTk, WTq, WTv, Kh, Qh, Vt);
  }
  // Kernel 2: flash attention (512 query tiles / 4 waves per block)
  {
    const int blocks = (BATCH * (SEQ / 16)) / 4;   // 128
    attn_kernel<<<blocks, 128, 0, stream>>>(Qh, Kh, Vt, out);
  }
}
